// EchoSpike_72327249264728
// MI455X (gfx1250) — matrix operand order repaired and checked by reading, not compiled, not run on hardware
//
#include <hip/hip_runtime.h>
#include <hip/hip_bf16.h>

// ---------------- problem constants (from reference) ----------------
#define Tsz 100
#define Bsz 128
#define Nsz 2048
#define Hsz 2048
#define Lsz 3
#define BETA 0.95f
#define THRESH 1.0f

typedef __attribute__((ext_vector_type(16))) _Float16 v16h;
typedef __attribute__((ext_vector_type(8)))  _Float16 v8h;
typedef __attribute__((ext_vector_type(8)))  float    v8f;

struct h8x2 { v8h lo, hi; };

static __device__ __forceinline__ v16h pack16(v8h lo, v8h hi) {
    h8x2 t{lo, hi};
    return __builtin_bit_cast(v16h, t);
}

// ---------------- helper kernels ----------------
__global__ void f32_to_f16_kernel(const float* __restrict__ in,
                                  _Float16* __restrict__ out, int n) {
    int i = blockIdx.x * blockDim.x + threadIdx.x;
    int stride = gridDim.x * blockDim.x;
    for (; i < n; i += stride) out[i] = (_Float16)in[i];
}

__global__ void zero_f32_kernel(float* __restrict__ p, int n) {
    int i = blockIdx.x * blockDim.x + threadIdx.x;
    int stride = gridDim.x * blockDim.x;
    for (; i < n; i += stride) p[i] = 0.0f;
}

// ---------------- fused WMMA GEMM + LIF + losses + trace ----------------
// One wave computes one 16x16 output tile of cur = layer_in @ W^T,
// then performs the Leaky (subtract-reset) update, spike threshold,
// loss partial reductions, running-mean spike-trace update, and emits
// the spikes as f16 for the next layer's A matrix.
__global__ void __launch_bounds__(256)
lif_layer_kernel(const _Float16* __restrict__ Ain,   // [B][N] layer input, f16
                 const _Float16* __restrict__ Wh,    // [H][N] layer weight, f16
                 float* __restrict__ mems,           // [B][H] membrane state
                 float* __restrict__ spk_tr,         // [B][H] running-mean trace
                 const float* __restrict__ prev_tr,  // [B][H] prev segment trace
                 _Float16* __restrict__ spk_out,     // [B][H] spikes (f16)
                 float* __restrict__ mem_out,        // [B][H] membranes to d_out (or null)
                 float* __restrict__ loss_out,       // {loss_pred, loss_con}
                 float tstep) {
    const int lane  = threadIdx.x & 31;
    const int wave  = threadIdx.x >> 5;
    const int hTile = blockIdx.x * 8 + wave;   // 0..127
    const int bTile = blockIdx.y;              // 0..7
    const int half  = lane >> 4;               // 0: lanes 0-15, 1: lanes 16-31
    const int ln    = lane & 15;

    // A fragment (16x32 f16): lane holds row m = ln.
    //   elems 0..7  -> K = kBase + half*8  + 0..7
    //   elems 8..15 -> K = kBase + 16 + half*8 + 0..7
    const _Float16* arow = Ain + (size_t)(bTile * 16 + ln) * Nsz + half * 8;
    // B fragment (32x16 f16): B[k][n] = W[hBase+n][k]; lane holds n = ln,
//   elems 0..7  -> K = kBase + half*8  + 0..7
    //   elems 8..15 -> K = kBase + 16 + half*8 + 0..7
    const _Float16* brow = Wh + (size_t)(hTile * 16 + ln) * Nsz + half * 8;

    v8f acc = {};

    // Software-pipelined K loop: load fragment k+32 before WMMA on fragment k,
    // so global_load_b128s overlap the matrix op instead of draining loadcnt.
    v8h a_lo = *(const v8h*)(arow);
    v8h a_hi = *(const v8h*)(arow + 16);
    v8h b_lo = *(const v8h*)(brow);
    v8h b_hi = *(const v8h*)(brow + 16);

    for (int k = 0; k < Nsz - 32; k += 32) {
        // Unconditional speculative prefetch of the streaming W row; always
        // lands inside the workspace allocation (state arrays follow W).
        __builtin_prefetch(brow + k + 256, 0, 3);  // global_prefetch_b8

        v8h na_lo = *(const v8h*)(arow + k + 32);
        v8h na_hi = *(const v8h*)(arow + k + 48);
        v8h nb_lo = *(const v8h*)(brow + k + 32);
        v8h nb_hi = *(const v8h*)(brow + k + 48);

        acc = __builtin_amdgcn_wmma_f32_16x16x32_f16(
            /*neg_a=*/false, pack16(a_lo, a_hi),
            /*neg_b=*/false, pack16(b_lo, b_hi),
            /*c_mod=*/(short)0, acc, /*reuse_a=*/false, /*reuse_b=*/false);

        a_lo = na_lo; a_hi = na_hi; b_lo = nb_lo; b_hi = nb_hi;
    }
    acc = __builtin_amdgcn_wmma_f32_16x16x32_f16(
        false, pack16(a_lo, a_hi), false, pack16(b_lo, b_hi),
        (short)0, acc, false, false);

    // C/D layout: VGPR i, lanes 0-15 -> M=i, N=lane; lanes 16-31 -> M=i+8, N=lane-16
    const int h = hTile * 16 + ln;
    float lp = 0.0f, lc = 0.0f;
#pragma unroll
    for (int i = 0; i < 8; ++i) {
        const int bb = bTile * 16 + half * 8 + i;
        const size_t idx = (size_t)bb * Hsz + h;
        const float old_mem = mems[idx];
        const float reset = (old_mem >= THRESH) ? 1.0f : 0.0f;
        const float mem = BETA * old_mem + acc[i] - reset * THRESH;
        const float spk = (mem >= THRESH) ? 1.0f : 0.0f;
        const float st = spk_tr[idx];          // trace BEFORE update (per reference)
        lp -= spk * st;
        lc += spk * prev_tr[idx];
        spk_tr[idx] = (st * tstep + spk) / (tstep + 1.0f);
        mems[idx] = mem;
        if (mem_out) mem_out[idx] = mem;
        spk_out[idx] = (_Float16)spk;
    }

    // wave32 tree reduction, then one atomic per wave
    for (int off = 16; off > 0; off >>= 1) {
        lp += __shfl_xor(lp, off, 32);
        lc += __shfl_xor(lc, off, 32);
    }
    if (lane == 0) {
        atomicAdd(&loss_out[0], lp * (1.0f / (float)Bsz));
        atomicAdd(&loss_out[1], lc * (1.0f / (float)Bsz));
    }
}

// ---------------- host-side orchestration ----------------
extern "C" void kernel_launch(void* const* d_in, const int* in_sizes, int n_in,
                              void* d_out, int out_size, void* d_ws, size_t ws_size,
                              hipStream_t stream) {
    const float* inp  = (const float*)d_in[0];  // [T,B,N]
    const float* W    = (const float*)d_in[1];  // [L,H,N]
    const float* prev = (const float*)d_in[2];  // [L,B,H]
    float* out = (float*)d_out;                 // [T,B,H] mems_T  ++  [T,L,2] losses

    // workspace layout
    char* ws = (char*)d_ws;
    size_t off = 0;
    _Float16* Wh = (_Float16*)(ws + off);  off += (size_t)Lsz * Hsz * Nsz * sizeof(_Float16);
    float* mems  = (float*)(ws + off);     off += (size_t)Lsz * Bsz * Hsz * sizeof(float);
    float* strc  = (float*)(ws + off);     off += (size_t)Lsz * Bsz * Hsz * sizeof(float);
    _Float16* inh = (_Float16*)(ws + off); off += (size_t)Bsz * Nsz * sizeof(_Float16);
    _Float16* s0 = (_Float16*)(ws + off);  off += (size_t)Bsz * Hsz * sizeof(_Float16);
    _Float16* s1 = (_Float16*)(ws + off);  off += (size_t)Bsz * Hsz * sizeof(_Float16);

    // one-time per call: convert weights, zero state + losses
    f32_to_f16_kernel<<<4096, 256, 0, stream>>>(W, Wh, Lsz * Hsz * Nsz);
    zero_f32_kernel<<<1024, 256, 0, stream>>>(mems, Lsz * Bsz * Hsz);
    zero_f32_kernel<<<1024, 256, 0, stream>>>(strc, Lsz * Bsz * Hsz);
    zero_f32_kernel<<<4, 256, 0, stream>>>(out + (size_t)Tsz * Bsz * Hsz, Tsz * Lsz * 2);

    const size_t BH = (size_t)Bsz * Hsz;
    const size_t HN = (size_t)Hsz * Nsz;
    dim3 grid(Hsz / 128, Bsz / 16);  // 16 x 8 blocks, 8 waves/block -> 1024 tiles

    for (int t = 0; t < Tsz; ++t) {
        f32_to_f16_kernel<<<256, 256, 0, stream>>>(inp + (size_t)t * Bsz * Nsz, inh,
                                                   Bsz * Nsz);
        float* losses = out + (size_t)Tsz * Bsz * Hsz + (size_t)t * Lsz * 2;
        const float tf = (float)t;

        // layer 0: in = inp[t], spikes -> s0
        lif_layer_kernel<<<grid, 256, 0, stream>>>(
            inh, Wh, mems, strc, prev, s0, nullptr, losses + 0, tf);
        // layer 1: in = s0, spikes -> s1
        lif_layer_kernel<<<grid, 256, 0, stream>>>(
            s0, Wh + HN, mems + BH, strc + BH, prev + BH, s1, nullptr, losses + 2, tf);
        // layer 2: in = s1, spikes -> s0 (unused), membranes -> d_out[t]
        lif_layer_kernel<<<grid, 256, 0, stream>>>(
            s1, Wh + 2 * HN, mems + 2 * BH, strc + 2 * BH, prev + 2 * BH, s0,
            out + (size_t)t * Bsz * Hsz, losses + 4, tf);
    }
}
